// EmbeddingMul_13520557048524
// MI455X (gfx1250) — compile-verified, matrix operand order checked by reading, not run on hardware
//
#include <hip/hip_runtime.h>

typedef __attribute__((ext_vector_type(16))) __bf16 v16bf;
typedef __attribute__((ext_vector_type(8)))  float  v8f;

#define NUM_EMB 32000
#define EMB_DIM 128
#define BATCH   8192
#define EPS     1e-4f
#define KITERS  500            // 32000 / 64

// Lane-invariant part of the K-offset pattern for 16-bit (bf16) A/B fragments
// of V_WMMA_*_16X16X32: element j of the per-lane v16bf maps to
//   k_local = cpat(j) + (lane >= 16 ? 8 : 0)
__device__ __forceinline__ int cpat(int j) {
  int r = j >> 1;
  return ((r < 4) ? 0 : 16) + ((r & 3) << 1) + (j & 1);
}

// stage 4 fragment-lanes (2 K-slices x 2 n-tiles) worth of weights, f32 -> bf16
__device__ __forceinline__ void load_frags(const float* __restrict__ wp,
                                           v16bf& fA0, v16bf& fB0,
                                           v16bf& fA1, v16bf& fB1) {
#pragma unroll
  for (int j = 0; j < 16; ++j) {
    fA0[j] = (__bf16)wp[cpat(j) * EMB_DIM];             // K-slice 0, n-tile wave
    fB0[j] = (__bf16)wp[cpat(j) * EMB_DIM + 64];        // K-slice 0, n-tile wave+4
    fA1[j] = (__bf16)wp[(cpat(j) + 32) * EMB_DIM];      // K-slice 1, n-tile wave
    fB1[j] = (__bf16)wp[(cpat(j) + 32) * EMB_DIM + 64]; // K-slice 1, n-tile wave+4
  }
}

__global__ __launch_bounds__(128, 1)
void embedmul_wmma_kernel(const float* __restrict__ x,
                          const float* __restrict__ W,
                          float* __restrict__ out) {
  // 32 KB shared: during the loop, bf16 B tiles for two 32-K slices (16 KB);
  // after the loop, reused as the f32 cross-wave reduction buffer (32 KB).
  __shared__ __attribute__((aligned(32))) unsigned char smem[32768];
  __bf16 (*ldsB)[8][32][16] = (__bf16 (*)[8][32][16])smem;  // [k-slice][n_tile][lane][j]
  v8f (*red)[32]            = (v8f (*)[32])smem;            // [mi*16 + frag][lane]

  const int tid   = threadIdx.x;
  const int lane  = tid & 31;
  const int wave  = tid >> 5;      // 0..3
  const int l16   = lane & 15;
  const int half  = lane >> 4;     // 16-lane half
  const int kbase = half << 3;     // +8 K-offset for lanes 16..31

  const int mi = wave & 1;         // row half of the 64-row block tile
  const int kh = wave >> 1;        // which 32-K slice of each 64-K chunk
  const int block_m = blockIdx.x * 64;

  // rows of this wave's two A fragments (both lane halves carry the same rows)
  const int m0 = block_m + mi * 32 + l16;
  const float x0 = x[m0];
  const float x1 = x[m0 + 16];
  const float xm2_0 = x0 * x0 + EPS;   // x^2 + eps
  const float xm2_1 = x1 * x1 + EPS;

  v8f acc0[8] = {};   // rows m0..+15 x 8 n-tiles (partial over this wave's K slices)
  v8f acc1[8] = {};   // rows +16

  // staging role: this thread supplies B-fragment lanes (both K-slices) for
  // n-tiles (wave) and (wave+4)
  const int ncolA = wave * 16 + l16;

  v16bf fA0, fB0, fA1, fB1;
  load_frags(W + (size_t)kbase * EMB_DIM + ncolA, fA0, fB0, fA1, fB1);

  for (int it = 0; it < KITERS; ++it) {
    __syncthreads();                                  // previous chunk's readers done
    *(v16bf*)&ldsB[0][wave][lane][0]     = fA0;
    *(v16bf*)&ldsB[0][wave + 4][lane][0] = fB0;
    *(v16bf*)&ldsB[1][wave][lane][0]     = fA1;
    *(v16bf*)&ldsB[1][wave + 4][lane][0] = fB1;
    __syncthreads();                                  // current chunk visible

    // issue next chunk's global loads early (overlap with WMMA below)
    {
      const int itn = (it + 1 < KITERS) ? it + 1 : KITERS - 1;   // tail reload
      load_frags(W + ((size_t)itn * 64 + kbase) * EMB_DIM + ncolA,
                 fA0, fB0, fA1, fB1);
    }

    // synthesize this wave's two A (score) fragments for its K slice:
    // d/eps = rcp(x^2 + eps - k^2); bf16 keeps the f32 exponent range
    v16bf a0, a1;
    const float kb = (float)(it * 64 + kh * 32 + kbase);
#pragma unroll
    for (int j = 0; j < 16; ++j) {
      float kf = kb + (float)cpat(j);
      float kk = kf * kf;
      a0[j] = (__bf16)__builtin_amdgcn_rcpf(xm2_0 - kk);
      a1[j] = (__bf16)__builtin_amdgcn_rcpf(xm2_1 - kk);
    }

    // all 8 n-tiles: one A pair feeds 16 WMMAs
#pragma unroll
    for (int t = 0; t < 8; ++t) {
      v16bf b = *(const v16bf*)&ldsB[kh][t][lane][0];
      acc0[t] = __builtin_amdgcn_wmma_f32_16x16x32_bf16(false, a0, false, b,
                                                        (short)0, acc0[t], false, false);
      acc1[t] = __builtin_amdgcn_wmma_f32_16x16x32_bf16(false, a1, false, b,
                                                        (short)0, acc1[t], false, false);
    }
  }

  // ---- cross-wave K reduction (2-way, deterministic), then scale by eps ----
  __syncthreads();                 // all ldsB readers done before smem reuse
  if (kh == 1) {
#pragma unroll
    for (int t = 0; t < 8; ++t) {
      red[mi * 16 + t * 2 + 0][lane] = acc0[t];
      red[mi * 16 + t * 2 + 1][lane] = acc1[t];
    }
  }
  __syncthreads();
  if (kh == 0) {
    // 16x16 f32 C/D layout: lanes 0-15: VGPR j -> M=j ; lanes 16-31: M=8+j
    const int row0 = block_m + mi * 32 + half * 8;
#pragma unroll
    for (int t = 0; t < 8; ++t) {
      v8f p0 = red[mi * 16 + t * 2 + 0][lane];
      v8f p1 = red[mi * 16 + t * 2 + 1][lane];
      const int ncol = t * 16 + l16;
#pragma unroll
      for (int j = 0; j < 8; ++j) {
        out[(size_t)(row0 + j)      * EMB_DIM + ncol] = EPS * (acc0[t][j] + p0[j]);
        out[(size_t)(row0 + 16 + j) * EMB_DIM + ncol] = EPS * (acc1[t][j] + p1[j]);
      }
    }
  }
}

extern "C" void kernel_launch(void* const* d_in, const int* in_sizes, int n_in,
                              void* d_out, int out_size, void* d_ws, size_t ws_size,
                              hipStream_t stream) {
  const float* x = (const float*)d_in[0];   // (8192,) f32
  const float* W = (const float*)d_in[1];   // (32000, 128) f32
  float* out = (float*)d_out;               // (8192, 128) f32

  dim3 grid(BATCH / 64);   // 128 blocks, each 64 rows x all 128 cols
  dim3 block(128);         // 4 waves: 2 row-halves x 2 K-slices, 32x128 wave tiles
  hipLaunchKernelGGL(embedmul_wmma_kernel, grid, block, 0, stream, x, W, out);
}
